// MultiHeadAttention_8065948582287
// MI455X (gfx1250) — matrix operand order repaired and checked by reading, not compiled, not run on hardware
//
#include <hip/hip_runtime.h>

typedef __attribute__((ext_vector_type(16))) _Float16 v16h;
typedef __attribute__((ext_vector_type(8)))  float    v8f;
typedef __attribute__((ext_vector_type(4)))  unsigned int u32x4;
typedef __attribute__((ext_vector_type(8)))  int      i32x8;
typedef __attribute__((ext_vector_type(4)))  int      i32x4;

constexpr int NB   = 4;
constexpr int SEQ  = 2048;
constexpr int EMB  = 1024;
constexpr int NH   = 16;
constexpr int HD   = 64;        // head dim
constexpr int MROWS = NB * SEQ; // 8192

// ---------------------------------------------------------------------------
// Tensor Data Mover: 2D f16 tile -> LDS. D# per CDNA5 ISA
// §8.3/8.4: group0 {count=1, lds_addr, global_addr, type=2}, group1
// {data_size=2B, tensor_dim0/1, tile_dim0/1, tensor_dim0_stride}; groups 2/3
// zero (<=2D tensor). This toolchain exposes the 6-arg builtin
// (g0 u32x4, g1 i32x8, g2 i32x4, g3 i32x4, ext i32x8, cpol). TENSORcnt-tracked.
// ---------------------------------------------------------------------------
__device__ __forceinline__ unsigned lds_off(const void* p) {
  // generic pointer to LDS: addr[31:0] is the LDS byte offset (aperture in MSBs)
  return (unsigned)(unsigned long long)p;
}

__device__ __forceinline__ void tdm_load_2d_f16(unsigned lds_addr, const void* gaddr,
                                                unsigned dim0, unsigned dim1,
                                                unsigned tile0, unsigned tile1,
                                                unsigned stride0) {
  const unsigned long long ga = (unsigned long long)gaddr;
  u32x4 g0;
  g0[0] = 1u;                                            // count=1, user mode
  g0[1] = lds_addr;                                      // LDS byte address
  g0[2] = (unsigned)ga;                                  // global_addr[31:0]
  g0[3] = (unsigned)((ga >> 32) & 0x01FFFFFFu) | (2u << 30); // [120:64] + type=2
  i32x8 g1;
  g1[0] = (int)(1u << 16);                               // data_size=1 (2 bytes)
  g1[1] = (int)((dim0 & 0xFFFFu) << 16);                 // tensor_dim0[15:0]
  g1[2] = (int)((dim0 >> 16) | ((dim1 & 0xFFFFu) << 16));// dim0[31:16] | dim1[15:0]
  g1[3] = (int)((dim1 >> 16) | (tile0 << 16));           // dim1[31:16] | tile_dim0
  g1[4] = (int)(tile1 & 0xFFFFu);                        // tile_dim1 (tile_dim2=0)
  g1[5] = (int)stride0;                                  // tensor_dim0_stride[31:0]
  g1[6] = 0;                                             // stride0[47:32]|stride1
  g1[7] = 0;
  const i32x4 z4 = {0, 0, 0, 0};
  const i32x8 z8 = {0, 0, 0, 0, 0, 0, 0, 0};
  __builtin_amdgcn_tensor_load_to_lds(g0, g1, z4, z4, z8, 0);
}

// ---------------------------------------------------------------------------
// Projection GEMM: P = X(f32)[8192x1024] @ W(f32)[1024x1024] -> f16 workspace
// vmode 0: dst[n][h][s][d]   vmode 1: dst[n][h][d][s] (V transposed)
// ---------------------------------------------------------------------------
__global__ __launch_bounds__(128) void proj_kernel(const float* __restrict__ X,
                                                   const float* __restrict__ W,
                                                   _Float16* __restrict__ dst,
                                                   float scale, int vmode) {
  const int lane = threadIdx.x & 31;
  const int wave = threadIdx.x >> 5;
  const int waveId = blockIdx.x * 4 + wave;
  const int mt = waveId >> 4;   // 0..511 row tiles
  const int nt = waveId & 15;   // 0..15  (64-wide col tile)
  const int c = lane & 15;
  const int half = lane >> 4;
  const int row = mt * 16 + c;  // A-frag row for this lane

  v8f acc[4] = {};
  for (int kk = 0; kk < EMB; kk += 32) {
    if (kk + 32 < EMB)  // pull next weight panel toward the caches
      __builtin_prefetch(W + (size_t)(kk + 32 + half * 16) * EMB + nt * 64 + c, 0, 0);
    // A fragment (16x32 f16): rows mt*16..+15, cols kk..kk+31, from f32
    v16h a;
    const float* xr = X + (size_t)row * EMB + kk + half * 16;
#pragma unroll
    for (int p = 0; p < 8; ++p) {
      const int kb = 2 * p;
      a[2 * p]     = (_Float16)xr[kb];
      a[2 * p + 1] = (_Float16)xr[kb + 1];
    }
#pragma unroll
    for (int ct = 0; ct < 4; ++ct) {
      const int col = nt * 64 + ct * 16 + c;
      v16h b;
      const float* wc = W + (size_t)(kk + half * 16) * EMB + col;
#pragma unroll
      for (int p = 0; p < 8; ++p) {
        b[2 * p]     = (_Float16)wc[(size_t)(2 * p) * EMB];
        b[2 * p + 1] = (_Float16)wc[(size_t)(2 * p + 1) * EMB];
      }
      acc[ct] = __builtin_amdgcn_wmma_f32_16x16x32_f16(
          false, a, false, b, (short)0, acc[ct], false, false);
    }
  }
  // scatter to per-head workspace layout
#pragma unroll
  for (int ct = 0; ct < 4; ++ct) {
    const int col = nt * 64 + ct * 16 + c;
    const int h = col >> 6, d = col & 63;
#pragma unroll
    for (int p = 0; p < 8; ++p) {
      const int m = p + half * 8;
      const int rg = mt * 16 + m;
      const int n = rg >> 11, s = rg & 2047;
      size_t idx;
      if (vmode)
        idx = ((size_t)((n * NH + h) * HD + d)) * SEQ + s;   // [n][h][d][s]
      else
        idx = ((size_t)((n * NH + h) * SEQ + s)) * HD + d;   // [n][h][s][d]
      dst[idx] = (_Float16)(acc[ct][p] * scale);
    }
  }
}

// ---------------------------------------------------------------------------
// Flash attention: one wave per (n, h, 16-row q tile). K/V chunks streamed
// into LDS double-buffers by the Tensor Data Mover, synchronized with
// s_wait_tensorcnt. Online softmax via LDS scratch; O in f32 C-fragments.
// Q already carries the 1/sqrt(D) scale.
// ---------------------------------------------------------------------------
__global__ __launch_bounds__(32) void attn_kernel(const _Float16* __restrict__ Q,
                                                  const _Float16* __restrict__ K,
                                                  const _Float16* __restrict__ V,
                                                  _Float16* __restrict__ O) {
  __shared__ _Float16 ldsK[2][32 * 64];   // [s_local][d]  (4KB each)
  __shared__ _Float16 ldsV[2][64 * 32];   // [d][s_local]  (4KB each)
  __shared__ float    ldsS[16 * 32];
  __shared__ _Float16 ldsP[16 * 32];
  __shared__ float rmax[16], rsum[16], rscale[16];

  const int lane = threadIdx.x & 31;
  const int c = lane & 15;
  const int half = lane >> 4;
  const int qt = blockIdx.x & 127;
  const int h  = (blockIdx.x >> 7) & 15;
  const int n  = blockIdx.x >> 11;

  const _Float16* qp = Q + ((size_t)(n * NH + h) * SEQ + qt * 16) * HD;
  const _Float16* kp = K + (size_t)(n * NH + h) * SEQ * HD;
  const _Float16* vp = V + (size_t)(n * NH + h) * HD * SEQ;

  if (lane < 16) { rmax[lane] = -1e30f; rsum[lane] = 0.0f; }
  __syncthreads();

  // Prime the TDM pipeline: first K (32x64, row stride 64) and V (64x32 slice
  // of the [64 x 2048] transposed tensor, row stride 2048) chunks.
  tdm_load_2d_f16(lds_off(&ldsK[0][0]), kp, HD, SEQ, HD, 32, HD);
  tdm_load_2d_f16(lds_off(&ldsV[0][0]), vp, SEQ, HD, 32, HD, SEQ);
  __builtin_amdgcn_s_wait_tensorcnt(0);

  // Q tile 16x64 as two A-fragments (d-chunks 0..31 and 32..63); loaded once.
  v16h aq[2];
#pragma unroll
  for (int ch = 0; ch < 2; ++ch) {
    const _Float16* qr = qp + c * HD + ch * 32 + half * 16;
#pragma unroll
    for (int p = 0; p < 8; ++p) {
      const int kb = 2 * p;
      aq[ch][2 * p]     = qr[kb];
      aq[ch][2 * p + 1] = qr[kb + 1];
    }
  }

  v8f o[4] = {};
  int cur = 0;
  for (int kb = 0; kb < SEQ; kb += 32) {
    const bool more = (kb + 32) < SEQ;
    if (more) {  // async-fetch next chunk into the other buffer
      tdm_load_2d_f16(lds_off(&ldsK[cur ^ 1][0]), kp + (size_t)(kb + 32) * HD,
                      HD, SEQ, HD, 32, HD);
      tdm_load_2d_f16(lds_off(&ldsV[cur ^ 1][0]), vp + (kb + 32),
                      SEQ, HD, 32, HD, SEQ);
    }
    // ---- S = Q @ K^T for 32 key positions (two 16x16 tiles) ----
    v8f s[2] = {};
#pragma unroll
    for (int t = 0; t < 2; ++t) {
#pragma unroll
      for (int ch = 0; ch < 2; ++ch) {
        v16h b;  // B = K^T chunk from LDS: rows = d, cols = key position
        const _Float16* kc = &ldsK[cur][(t * 16 + c) * HD + ch * 32 + half * 16];
#pragma unroll
        for (int p = 0; p < 8; ++p) {
          b[2 * p]     = kc[2 * p];
          b[2 * p + 1] = kc[2 * p + 1];
        }
        s[t] = __builtin_amdgcn_wmma_f32_16x16x32_f16(
            false, aq[ch], false, b, (short)0, s[t], false, false);
      }
    }
    // ---- stash scores to LDS ----
#pragma unroll
    for (int t = 0; t < 2; ++t)
#pragma unroll
      for (int p = 0; p < 8; ++p)
        ldsS[(p + half * 8) * 32 + t * 16 + c] = s[t][p];
    __syncthreads();

    // ---- online softmax: lanes 0..15 own one q-row each ----
    if (lane < 16) {
      const int r = lane;
      const float mo = rmax[r];
      float mx = mo;
#pragma unroll 4
      for (int i = 0; i < 32; ++i) mx = fmaxf(mx, ldsS[r * 32 + i]);
      const float sc = __expf(mo - mx);
      float su = rsum[r] * sc;
#pragma unroll 4
      for (int i = 0; i < 32; ++i) {
        const float e = __expf(ldsS[r * 32 + i] - mx);
        su += e;
        ldsP[r * 32 + i] = (_Float16)e;
      }
      rmax[r] = mx; rsum[r] = su; rscale[r] = sc;
    }
    __syncthreads();

    // ---- rescale running O by exp(m_old - m_new) per row ----
#pragma unroll
    for (int p = 0; p < 8; ++p) {
      const float sc = rscale[p + half * 8];
#pragma unroll
      for (int ct = 0; ct < 4; ++ct) o[ct][p] *= sc;
    }
    // ---- P A-fragment from LDS ----
    v16h ap;
#pragma unroll
    for (int p = 0; p < 8; ++p) {
      const int kbp = 2 * p + half * 16;
      ap[2 * p]     = ldsP[c * 32 + kbp];
      ap[2 * p + 1] = ldsP[c * 32 + kbp + 1];
    }
    // ---- O += P @ V_chunk (LDS V stored [d][s_local]) ----
#pragma unroll
    for (int ct = 0; ct < 4; ++ct) {
      v16h b;
      const _Float16* vc = &ldsV[cur][(ct * 16 + c) * 32 + half * 16];
#pragma unroll
      for (int p = 0; p < 8; ++p) {
        b[2 * p]     = vc[2 * p];
        b[2 * p + 1] = vc[2 * p + 1];
      }
      o[ct] = __builtin_amdgcn_wmma_f32_16x16x32_f16(
          false, ap, false, b, (short)0, o[ct], false, false);
    }
    if (more) __builtin_amdgcn_s_wait_tensorcnt(0);  // next buffer landed
    cur ^= 1;
  }

  // ---- normalize and write O as f16 row-major [N*S, E] ----
#pragma unroll
  for (int p = 0; p < 8; ++p) {
    const int m = p + half * 8;
    const float inv = 1.0f / rsum[m];
    const int srow = qt * 16 + m;
#pragma unroll
    for (int ct = 0; ct < 4; ++ct)
      O[((size_t)(n * SEQ + srow)) * EMB + h * HD + ct * 16 + c] =
          (_Float16)(o[ct][p] * inv);
  }
}

// ---------------------------------------------------------------------------
// Output GEMM: out = O(f16)[8192x1024] @ Wo(f32) + bo, f32 result.
// ---------------------------------------------------------------------------
__global__ __launch_bounds__(128) void oproj_kernel(const _Float16* __restrict__ A,
                                                    const float* __restrict__ W,
                                                    const float* __restrict__ bias,
                                                    float* __restrict__ out) {
  const int lane = threadIdx.x & 31;
  const int wave = threadIdx.x >> 5;
  const int waveId = blockIdx.x * 4 + wave;
  const int mt = waveId >> 4;
  const int nt = waveId & 15;
  const int c = lane & 15;
  const int half = lane >> 4;
  const int row = mt * 16 + c;

  v8f acc[4];
#pragma unroll
  for (int ct = 0; ct < 4; ++ct) {
    const float bb = bias[nt * 64 + ct * 16 + c];
#pragma unroll
    for (int p = 0; p < 8; ++p) acc[ct][p] = bb;
  }

  for (int kk = 0; kk < EMB; kk += 32) {
    if (kk + 32 < EMB)
      __builtin_prefetch(W + (size_t)(kk + 32 + half * 16) * EMB + nt * 64 + c, 0, 0);
    v16h a;
    const _Float16* ar = A + (size_t)row * EMB + kk + half * 16;
#pragma unroll
    for (int p = 0; p < 8; ++p) {
      const int kb = 2 * p;
      a[2 * p]     = ar[kb];
      a[2 * p + 1] = ar[kb + 1];
    }
#pragma unroll
    for (int ct = 0; ct < 4; ++ct) {
      const int col = nt * 64 + ct * 16 + c;
      v16h b;
      const float* wc = W + (size_t)(kk + half * 16) * EMB + col;
#pragma unroll
      for (int p = 0; p < 8; ++p) {
        b[2 * p]     = (_Float16)wc[(size_t)(2 * p) * EMB];
        b[2 * p + 1] = (_Float16)wc[(size_t)(2 * p + 1) * EMB];
      }
      acc[ct] = __builtin_amdgcn_wmma_f32_16x16x32_f16(
          false, a, false, b, (short)0, acc[ct], false, false);
    }
  }
#pragma unroll
  for (int ct = 0; ct < 4; ++ct) {
    const int col = nt * 64 + ct * 16 + c;
#pragma unroll
    for (int p = 0; p < 8; ++p) {
      const int rg = mt * 16 + p + half * 8;
      out[(size_t)rg * EMB + col] = acc[ct][p];
    }
  }
}

// ---------------------------------------------------------------------------
extern "C" void kernel_launch(void* const* d_in, const int* in_sizes, int n_in,
                              void* d_out, int out_size, void* d_ws, size_t ws_size,
                              hipStream_t stream) {
  const float* queries = (const float*)d_in[0];
  const float* keys    = (const float*)d_in[1];
  const float* values  = (const float*)d_in[2];
  const float* Wq      = (const float*)d_in[3];
  const float* Wk      = (const float*)d_in[4];
  const float* Wv      = (const float*)d_in[5];
  const float* Wo      = (const float*)d_in[6];
  const float* bo      = (const float*)d_in[7];
  (void)in_sizes; (void)n_in; (void)out_size; (void)ws_size;

  // workspace partition: 4 x 16MB f16 buffers (Q, K, V^T, O)
  char* ws = (char*)d_ws;
  const size_t seg = (size_t)MROWS * EMB * sizeof(_Float16);  // 16 MB
  _Float16* q_ws = (_Float16*)(ws);
  _Float16* k_ws = (_Float16*)(ws + seg);
  _Float16* v_ws = (_Float16*)(ws + 2 * seg);
  _Float16* o_ws = (_Float16*)(ws + 3 * seg);

  const dim3 blk(128);
  const dim3 grd((MROWS / 16) * (EMB / 64) / 4);  // 2048 blocks, 4 waves each

  proj_kernel<<<grd, blk, 0, stream>>>(queries, Wq, q_ws, 0.125f, 0); // 1/sqrt(64)
  proj_kernel<<<grd, blk, 0, stream>>>(keys,    Wk, k_ws, 1.0f,   0);
  proj_kernel<<<grd, blk, 0, stream>>>(values,  Wv, v_ws, 1.0f,   1); // transposed

  attn_kernel<<<NB * NH * (SEQ / 16), 32, 0, stream>>>(q_ws, k_ws, v_ws, o_ws);

  oproj_kernel<<<grd, blk, 0, stream>>>(o_ws, Wo, bo, (float*)d_out);
}
